// NonMonotonicSTLM_82824149336839
// MI455X (gfx1250) — matrix operand order repaired and checked by reading, not compiled, not run on hardware
//
#include <hip/hip_runtime.h>

typedef unsigned short u16;
typedef unsigned int   u32;

typedef __attribute__((ext_vector_type(16))) __bf16 v16bf;
typedef __attribute__((ext_vector_type(8)))  float  v8f;
typedef __attribute__((ext_vector_type(4)))  u32    u32x4;

#define T_SEQ  512
#define BATCH  4
#define EDIM   1024
#define GDIM   4096      // 4*E
#define VOCAB  32000
#define ROWS   2048      // B*T
#define EOSIDX 31999
#define LOG_BOUND_F (-1.0005003335835335e-3f)

// ---------------- helpers ----------------
__device__ __forceinline__ u16 f2bf(float f) {
  u32 u = __float_as_uint(f);
  u32 r = ((u >> 16) & 1u) + 0x7fffu;      // round to nearest even
  return (u16)((u + r) >> 16);
}
__device__ __forceinline__ float bflo(u32 w) { return __uint_as_float(w << 16); }
__device__ __forceinline__ float bfhi(u32 w) { return __uint_as_float(w & 0xffff0000u); }
__device__ __forceinline__ float sigmoidf(float x) { return 1.0f / (1.0f + __expf(-x)); }
__device__ __forceinline__ float logsigf(float x) {
  return fminf(x, 0.0f) - log1pf(expf(-fabsf(x)));
}

// ---------------- elementwise / gather kernels ----------------
__global__ void k_f32_to_bf16(const float* __restrict__ in, u16* __restrict__ out, size_t n) {
  size_t i = (size_t)blockIdx.x * blockDim.x + threadIdx.x;
  size_t stride = (size_t)gridDim.x * blockDim.x;
  for (; i < n; i += stride) out[i] = f2bf(in[i]);
}

__global__ void k_bias_add(const float* __restrict__ a, const float* __restrict__ b,
                           float* __restrict__ out, int n) {
  int i = blockIdx.x * blockDim.x + threadIdx.x;
  if (i < n) out[i] = a[i] + b[i];
}

__global__ void k_embed_gather(const int* __restrict__ tokens, const float* __restrict__ emb,
                               u16* __restrict__ X) {
  size_t i = (size_t)blockIdx.x * blockDim.x + threadIdx.x;   // ROWS*EDIM threads
  int r = (int)(i >> 10);
  int e = (int)(i & (EDIM - 1));
  int tok = tokens[r];
  X[i] = f2bf(emb[(size_t)tok * EDIM + e]);
}

// ---------------- WMMA bf16 GEMM:  C(M,N) = A(M,K) * B(N,K)^T ----------------
// A row-major bf16, B row-major bf16 (weights as (N,K)), C row-major f32.
// Block tile = 64(M) x 256(N): 8 waves in a 2(M) x 4(N) grid, each wave
// computes 32x64 via 2 A frags x 4 B frags -> 8 WMMAs per K-step of 32.
// Requires M % 64 == 0 and N % 256 == 0 (true for all three GEMMs here), so
// EXEC is all-ones for every WMMA with no guards.
union FragAB { u32x4 q[2]; v16bf v; };

__device__ __forceinline__ v8f wmma_bf16(const FragAB& a, const FragAB& b, v8f c) {
  return __builtin_amdgcn_wmma_f32_16x16x32_bf16(false, a.v, false, b.v,
                                                 (short)0, c, false, false);
}

__global__ void k_gemm_bf16(const u16* __restrict__ A, const u16* __restrict__ Bm,
                            float* __restrict__ C, int M, int N, int K) {
  const int lane = threadIdx.x & 31;
  const int wave = threadIdx.x >> 5;
  const int r    = lane & 15;
  const int hi   = lane >> 4;            // 0 or 1
  const int wm   = wave >> 2;            // 0..1
  const int wn   = wave & 3;             // 0..3
  const int m0   = blockIdx.y * 64 + wm * 32;
  const int n0   = blockIdx.x * 256 + wn * 64;

  // A frag rows (this code's k order, not the ISA table order: lanes0-15 K=0..15,
  // lanes16-31 K=16..31, contiguous 16 elements per lane)
  const u16* arow0 = A + (size_t)(m0 + r) * K + hi * 16;
  const u16* arow1 = A + (size_t)(m0 + 16 + r) * K + hi * 16;
  // B frag rows (same k order as A: lanes0-15 K=0..15, lanes16-31 K=16..31,
  // contiguous 16 elements per lane)
  const u16* bp0 = Bm + (size_t)(n0      + r) * K + hi * 16;
  const u16* bp1 = Bm + (size_t)(n0 + 16 + r) * K + hi * 16;
  const u16* bp2 = Bm + (size_t)(n0 + 32 + r) * K + hi * 16;
  const u16* bp3 = Bm + (size_t)(n0 + 48 + r) * K + hi * 16;

  v8f acc[8];
#pragma unroll
  for (int u = 0; u < 8; ++u) acc[u] = (v8f){};

  for (int k0 = 0; k0 < K; k0 += 32) {
    FragAB a0, a1;
    a0.q[0] = *(const u32x4*)(arow0 + k0);
    a0.q[1] = *(const u32x4*)(arow0 + k0 + 8);
    a1.q[0] = *(const u32x4*)(arow1 + k0);
    a1.q[1] = *(const u32x4*)(arow1 + k0 + 8);

    FragAB f0, f1, f2, f3;
    f0.q[0] = *(const u32x4*)(bp0 + k0); f0.q[1] = *(const u32x4*)(bp0 + k0 + 8);
    f1.q[0] = *(const u32x4*)(bp1 + k0); f1.q[1] = *(const u32x4*)(bp1 + k0 + 8);
    f2.q[0] = *(const u32x4*)(bp2 + k0); f2.q[1] = *(const u32x4*)(bp2 + k0 + 8);
    f3.q[0] = *(const u32x4*)(bp3 + k0); f3.q[1] = *(const u32x4*)(bp3 + k0 + 8);

    acc[0] = wmma_bf16(a0, f0, acc[0]);
    acc[1] = wmma_bf16(a0, f1, acc[1]);
    acc[2] = wmma_bf16(a0, f2, acc[2]);
    acc[3] = wmma_bf16(a0, f3, acc[3]);
    acc[4] = wmma_bf16(a1, f0, acc[4]);
    acc[5] = wmma_bf16(a1, f1, acc[5]);
    acc[6] = wmma_bf16(a1, f2, acc[6]);
    acc[7] = wmma_bf16(a1, f3, acc[7]);
  }

  // C/D layout: element (m = vgpr + 8*hi, n = lane&15) within each 16x16 tile
#pragma unroll
  for (int u = 0; u < 4; ++u) {
    const int nn = n0 + u * 16 + r;
#pragma unroll
    for (int j = 0; j < 8; ++j) {
      C[(size_t)(m0 + hi * 8 + j) * N + nn]      = acc[u][j];
      C[(size_t)(m0 + 16 + hi * 8 + j) * N + nn] = acc[4 + u][j];
    }
  }
}

// ---------------- LSTM recurrent step ----------------
// One launch per timestep. 16 blocks x 256 threads. Block bk owns hidden
// columns j in [bk*64, bk*64+64) for ALL 4 gates and all 4 batch rows, so the
// nonlinear update needs no cross-block sync. h_prev is ping-ponged across
// steps to avoid a cross-block read/write race.
__global__ void k_lstm_step(const u16* __restrict__ Whh,   // (4096,1024) bf16
                            const float* __restrict__ xW,  // (2048,4096) f32, row = b*T+t
                            const float* __restrict__ bias,// 4096 f32  (bih+bhh)
                            const u16* __restrict__ hin,   // (4,1024) bf16
                            u16* __restrict__ hout,        // (4,1024) bf16
                            float* __restrict__ c,         // (4,1024) f32
                            u16* __restrict__ hall,        // (2048,1024) bf16
                            int t) {
  __shared__ float psum[4096];   // 1024 dots x 4 K-parts
  __shared__ u32   hsh[2048];    // 4x1024 bf16

  const int tid = threadIdx.x;
  const int bk  = blockIdx.x;
  const int j0  = bk * 64;

  const u32* hp = (const u32*)hin;
#pragma unroll
  for (int i = tid; i < 2048; i += 256) hsh[i] = hp[i];
  __syncthreads();

  // 1024 dots (4 gates x 64 cols x 4 batch), each split into 4 K-parts of 256
  for (int s = 0; s < 16; ++s) {
    int p      = tid + 256 * s;        // 0..4095
    int kpart  = p & 3;
    int dotIdx = p >> 2;               // 0..1023
    int b  = dotIdx & 3;
    int jj = (dotIdx >> 2) & 63;
    int gi = dotIdx >> 8;
    const u32* wr = (const u32*)(Whh + (size_t)(gi * EDIM + j0 + jj) * EDIM) + kpart * 128;
    const u32* hr = &hsh[b * 512 + kpart * 128];
    float acc = 0.0f;
#pragma unroll 4
    for (int k = 0; k < 128; ++k) {
      u32 w = wr[k], h2 = hr[k];
      acc = fmaf(bflo(w), bflo(h2), acc);
      acc = fmaf(bfhi(w), bfhi(h2), acc);
    }
    psum[p] = acc;
  }
  __syncthreads();

  // 256 updates: (b, jj)
  {
    int b  = tid & 3;
    int jj = tid >> 2;                 // 0..63
    int j  = j0 + jj;
    float g[4];
#pragma unroll
    for (int gi = 0; gi < 4; ++gi) {
      int dotIdx = gi * 256 + jj * 4 + b;
      float d = psum[dotIdx * 4] + psum[dotIdx * 4 + 1]
              + psum[dotIdx * 4 + 2] + psum[dotIdx * 4 + 3];
      int col = gi * EDIM + j;
      g[gi] = d + bias[col] + xW[(size_t)(b * T_SEQ + t) * GDIM + col];
    }
    float ci = c[b * EDIM + j];
    float iv = sigmoidf(g[0]);
    float fv = sigmoidf(g[1]);
    float gv = tanhf(g[2]);
    float ov = sigmoidf(g[3]);
    float cn = fv * ci + iv * gv;
    float hn = ov * tanhf(cn);
    c[b * EDIM + j] = cn;
    u16 hb = f2bf(hn);
    hout[b * EDIM + j] = hb;
    hall[(size_t)(b * T_SEQ + t) * EDIM + j] = hb;
  }
}

// ---------------- fused log-softmax + EOS-bound postprocess (in place) ------
// Key identity: after shift s = log_lb + log_eos_lb, logsumexp(v_lprobs) == s,
// so the final normalizer is Z = logaddexp(s, lprob_eos) -- a scalar.
__global__ void k_postprocess(float* __restrict__ out) {
  const int r   = blockIdx.x;         // 0..2047, r = b*T + t
  const int tid = threadIdx.x;        // 256
  float* row = out + (size_t)r * VOCAB;
  const int t = r & (T_SEQ - 1);

  __shared__ float red[256];

  float mx = -3.0e38f;
  for (int n = tid; n < EOSIDX; n += 256) mx = fmaxf(mx, row[n]);
  red[tid] = mx;
  __syncthreads();
  for (int s = 128; s > 0; s >>= 1) {
    if (tid < s) red[tid] = fmaxf(red[tid], red[tid + s]);
    __syncthreads();
  }
  mx = red[0];
  __syncthreads();

  float sum = 0.0f;
  for (int n = tid; n < EOSIDX; n += 256) sum += expf(row[n] - mx);
  red[tid] = sum;
  __syncthreads();
  for (int s = 128; s > 0; s >>= 1) {
    if (tid < s) red[tid] += red[tid + s];
    __syncthreads();
  }
  sum = red[0];

  float lse        = mx + logf(sum);
  float eos        = row[EOSIDX];
  float log_lb     = (float)(t + 1) * LOG_BOUND_F;
  float log_eos_lb = logsigf(-eos);
  float log_eos_ub = logsigf(eos);
  float s_shift    = log_lb + log_eos_lb;
  float log_lb_eos = logf(-expm1f(log_lb)) + log_eos_lb;
  float lprob_eos  = log_lb_eos - logsigf(log_lb_eos - log_eos_ub);
  float a  = fmaxf(s_shift, lprob_eos);
  float bm = fminf(s_shift, lprob_eos);
  float Z  = a + log1pf(expf(bm - a));

  float add = s_shift - lse - Z;
  for (int n = tid; n < EOSIDX; n += 256) row[n] = row[n] + add;
  if (tid == 0) row[EOSIDX] = lprob_eos - Z;
}

// ---------------- host orchestration ----------------
extern "C" void kernel_launch(void* const* d_in, const int* in_sizes, int n_in,
                              void* d_out, int out_size, void* d_ws, size_t ws_size,
                              hipStream_t stream) {
  (void)in_sizes; (void)n_in; (void)out_size; (void)ws_size;

  const int*   tokens = (const int*)  d_in[0];
  const float* emb    = (const float*)d_in[1];
  const float* Wproj  = (const float*)d_in[2];
  const float* Wih0   = (const float*)d_in[3];
  const float* Whh0   = (const float*)d_in[4];
  const float* bih0   = (const float*)d_in[5];
  const float* bhh0   = (const float*)d_in[6];
  const float* Wih1   = (const float*)d_in[7];
  const float* Whh1   = (const float*)d_in[8];
  const float* bih1   = (const float*)d_in[9];
  const float* bhh1   = (const float*)d_in[10];

  char* ws = (char*)d_ws;
  constexpr size_t SZ_W    = (size_t)GDIM * EDIM * 2;     // 8 MB
  constexpr size_t SZ_WP   = (size_t)VOCAB * EDIM * 2;    // 64 MB
  constexpr size_t SZ_BIAS = (size_t)GDIM * 4;            // 16 KB
  constexpr size_t SZ_X    = (size_t)ROWS * EDIM * 2;     // 4 MB
  constexpr size_t SZ_XW   = (size_t)ROWS * GDIM * 4;     // 32 MB
  constexpr size_t SZ_C    = (size_t)BATCH * EDIM * 4;    // 16 KB
  constexpr size_t SZ_HP   = (size_t)BATCH * EDIM * 2;    // 8 KB

  size_t off = 0;
  u16*   wih0b  = (u16*)  (ws + off); off += SZ_W;
  u16*   whh0b  = (u16*)  (ws + off); off += SZ_W;
  u16*   wih1b  = (u16*)  (ws + off); off += SZ_W;
  u16*   whh1b  = (u16*)  (ws + off); off += SZ_W;
  u16*   wprojb = (u16*)  (ws + off); off += SZ_WP;
  float* bias0  = (float*)(ws + off); off += SZ_BIAS;
  float* bias1  = (float*)(ws + off); off += SZ_BIAS;
  u16*   Xb     = (u16*)  (ws + off); off += SZ_X;
  u16*   Hb     = (u16*)  (ws + off); off += SZ_X;
  float* xW     = (float*)(ws + off); off += SZ_XW;
  float* cbuf   = (float*)(ws + off); off += SZ_C;
  u16*   hpA    = (u16*)  (ws + off); off += SZ_HP;
  u16*   hpB    = (u16*)  (ws + off); off += SZ_HP;

  // 1) weights -> bf16, bias combine, embedding gather (bf16)
  k_f32_to_bf16<<<8192, 256, 0, stream>>>(Wih0,  wih0b,  (size_t)GDIM * EDIM);
  k_f32_to_bf16<<<8192, 256, 0, stream>>>(Whh0,  whh0b,  (size_t)GDIM * EDIM);
  k_f32_to_bf16<<<8192, 256, 0, stream>>>(Wih1,  wih1b,  (size_t)GDIM * EDIM);
  k_f32_to_bf16<<<8192, 256, 0, stream>>>(Whh1,  whh1b,  (size_t)GDIM * EDIM);
  k_f32_to_bf16<<<8192, 256, 0, stream>>>(Wproj, wprojb, (size_t)VOCAB * EDIM);
  k_bias_add<<<GDIM / 256, 256, 0, stream>>>(bih0, bhh0, bias0, GDIM);
  k_bias_add<<<GDIM / 256, 256, 0, stream>>>(bih1, bhh1, bias1, GDIM);
  k_embed_gather<<<(ROWS * EDIM) / 256, 256, 0, stream>>>(tokens, emb, Xb);

  dim3 gGates(GDIM / 256, ROWS / 64);                 // (16, 32)
  dim3 gProj(VOCAB / 256, ROWS / 64);                 // (125, 32)

  // 2) layer 0: hoisted input GEMM, then recurrent steps
  hipMemsetAsync(cbuf, 0, SZ_C, stream);
  hipMemsetAsync(hpA, 0, SZ_HP, stream);
  k_gemm_bf16<<<gGates, 256, 0, stream>>>(Xb, wih0b, xW, ROWS, GDIM, EDIM);
  for (int t = 0; t < T_SEQ; ++t) {
    const u16* hin = (t & 1) ? hpB : hpA;
    u16*       ho  = (t & 1) ? hpA : hpB;
    k_lstm_step<<<16, 256, 0, stream>>>(whh0b, xW, bias0, hin, ho, cbuf, Hb, t);
  }

  // 3) layer 1 (input = h0_all, output overwrites Xb)
  hipMemsetAsync(cbuf, 0, SZ_C, stream);
  hipMemsetAsync(hpA, 0, SZ_HP, stream);
  k_gemm_bf16<<<gGates, 256, 0, stream>>>(Hb, wih1b, xW, ROWS, GDIM, EDIM);
  for (int t = 0; t < T_SEQ; ++t) {
    const u16* hin = (t & 1) ? hpB : hpA;
    u16*       ho  = (t & 1) ? hpA : hpB;
    k_lstm_step<<<16, 256, 0, stream>>>(whh1b, xW, bias1, hin, ho, cbuf, Xb, t);
  }

  // 4) projection GEMM straight into d_out, then in-place postprocess
  k_gemm_bf16<<<gProj, 256, 0, stream>>>(Xb, wprojb, (float*)d_out, ROWS, VOCAB, EDIM);
  k_postprocess<<<ROWS, 256, 0, stream>>>((float*)d_out);
}
